// EarthSpecificBlock_68470368633282
// MI455X (gfx1250) — compile-verified, matrix operand order checked by reading, not run on hardware
//
#include <hip/hip_runtime.h>

// ---------------------------------------------------------------------------
// Earth-specific transformer block (Pangu-style) for MI455X / gfx1250.
// All matmuls use v_wmma_f32_16x16x32_bf16 (wave32, 16x16 tiles, K=32).
// LDS staging uses CDNA5 async global->LDS loads (ASYNCcnt).
// ---------------------------------------------------------------------------

#define DIMC   192
#define HEADS  6
#define HD     32
#define NTOK   144            // tokens per window (2*6*12)
#define NWIN   64             // windows per lon-group (4*16)
#define NLONG  15             // lon groups
#define NBLK   (NLONG*NWIN)   // 960
#define HIDDEN 768
#define LTOK   131040         // 8*91*180
#define TBLK   910            // LTOK / 144

typedef __attribute__((ext_vector_type(16))) __bf16        v16bf;
typedef __attribute__((ext_vector_type(8)))  float         v8f;
typedef __attribute__((ext_vector_type(4)))  unsigned int  u32x4;

union Frag { v16bf v; u32x4 q[2]; };

__device__ __forceinline__ unsigned short f2bf(float f) {
  unsigned int u = __float_as_uint(f);
  u += 0x7fffu + ((u >> 16) & 1u);          // round-to-nearest-even
  return (unsigned short)(u >> 16);
}

__device__ __forceinline__ v8f vzero8() {
  v8f z = {0.f,0.f,0.f,0.f,0.f,0.f,0.f,0.f};
  return z;
}

// Fragment load per ISA 16-bit A/B layout: lane-half selects the +8 K offset;
// each lane reads two contiguous 8-element (16B) runs -> b128 loads.
__device__ __forceinline__ Frag ldfrag(const unsigned short* rowptr, int k0, int half) {
  Frag f;
  f.q[0] = *(const u32x4*)(rowptr + k0 + half * 8);
  f.q[1] = *(const u32x4*)(rowptr + k0 + 16 + half * 8);
  return f;
}

__device__ __forceinline__ v8f bwmma(Frag a, Frag b, v8f c) {
  return __builtin_amdgcn_wmma_f32_16x16x32_bf16(
      /*neg_a=*/false, a.v, /*neg_b=*/false, b.v,
      /*c_mod=*/(short)0, c, /*reuse_a=*/false, /*reuse_b=*/false);
}

// CDNA5 async global->LDS copy, 16B per lane. VDST VGPR carries the LDS byte
// offset (low 32 bits of the generic shared-pointer); VADDR the 64-bit global
// address. Tracked by ASYNCcnt.
__device__ __forceinline__ void async_copy_b128(const void* gsrc, void* lds_dst) {
  unsigned long long g = (unsigned long long)gsrc;
  unsigned int       l = (unsigned int)(unsigned long long)lds_dst;
  asm volatile("global_load_async_to_lds_b128 %0, %1, off"
               :: "v"(l), "v"(g) : "memory");
}

__device__ __forceinline__ void wait_async() {
  asm volatile("s_wait_asynccnt 0x0" ::: "memory");
}

// Shift-window region id for token n of window w in lon-group lonG.
__device__ __forceinline__ int region_id(int w, int lonG, int n) {
  int wp = n / 72, wa = (n / 12) % 6, wo = n % 12;
  int pl  = (w >> 4) * 2 + wp;
  int lat = (w & 15) * 6 + wa;
  int lon = lonG * 12 + wo;
  int rp = pl  < 6  ? 0 : (pl  == 6 ? 1 : 2);
  int ra = lat < 90 ? 0 : (lat < 93 ? 1 : 2);
  int ro = lon < 174 ? 0 : 1;
  return (rp * 3 + ra) * 3 + ro;
}

// ---------------------------------------------------------------------------
// Kernel W: weight fp32 -> bf16
// ---------------------------------------------------------------------------
__global__ void k_cvt(const float* __restrict__ src, unsigned short* __restrict__ dst, int n) {
  int i = blockIdx.x * 256 + threadIdx.x;
  if (i < n) dst[i] = f2bf(src[i]);
}

// ---------------------------------------------------------------------------
// Kernel A: LN1 + shifted-window gather + QKV GEMM (144x192 @ 192x576)
// qkv_out layout: [block(=lonG*64+w)][qkv][head][n][d]  (bf16)
// ---------------------------------------------------------------------------
__global__ __launch_bounds__(256) void k_ln1_qkv(
    const float* __restrict__ x,
    const float* __restrict__ n1w, const float* __restrict__ n1b,
    const unsigned short* __restrict__ wq, const float* __restrict__ qkvb,
    unsigned short* __restrict__ qkv_out)
{
  __shared__ unsigned short Ah[NTOK * DIMC];        // 55296 B, bf16 LN'ed tile
  const int blk  = blockIdx.x;                      // 0..959
  const int lonG = blk / NWIN, w = blk % NWIN;
  const int wave = threadIdx.x >> 5, lane = threadIdx.x & 31;

  // ---- gather + LayerNorm (fp32), one wave per token row ----
  for (int n = wave; n < NTOK; n += 8) {
    int wp = n / 72, wa = (n / 12) % 6, wo = n % 12;
    int plw  = (w >> 4) * 2 + wp;
    int latw = (w & 15) * 6 + wa;
    int lonw = lonG * 12 + wo;
    int plp  = (plw + 1) & 7;                       // undo roll (shift -1)
    int latp = latw + 3; if (latp >= 96)  latp -= 96;
    int lon  = lonw + 6; if (lon  >= 180) lon  -= 180;
    int lat  = latp - 2;
    if (lat < 0 || lat >= 91) {                     // lat padding -> zero row
      for (int kk = lane; kk < DIMC; kk += 32) Ah[n * DIMC + kk] = 0;
    } else {
      const float* xr = x + (((long)plp * 91 + lat) * 180 + lon) * DIMC;
      float vals[6], s = 0.f, s2 = 0.f;
      #pragma unroll
      for (int i = 0; i < 6; ++i) {
        float v = xr[lane + 32 * i]; vals[i] = v; s += v; s2 += v * v;
      }
      #pragma unroll
      for (int m = 16; m; m >>= 1) { s += __shfl_xor(s, m, 32); s2 += __shfl_xor(s2, m, 32); }
      float mean = s * (1.f / DIMC);
      float inv  = rsqrtf(s2 * (1.f / DIMC) - mean * mean + 1e-5f);
      #pragma unroll
      for (int i = 0; i < 6; ++i) {
        int kk = lane + 32 * i;
        Ah[n * DIMC + kk] = f2bf((vals[i] - mean) * inv * n1w[kk] + n1b[kk]);
      }
    }
  }
  __syncthreads();

  // ---- GEMM: 9 M-tiles x 36 N-tiles, K-loop of 6 ----
  const int half = lane >> 4, lm = lane & 15;
  for (int tile = wave; tile < 9 * 36; tile += 8) {
    int mi = tile % 9, nj = tile / 9;
    int m0 = mi * 16, n0 = nj * 16;
    v8f acc = vzero8();
    const unsigned short* arow = Ah + (m0 + lm) * DIMC;
    const unsigned short* brow = wq + (n0 + lm) * DIMC;    // qkv_w row j, K contiguous
    #pragma unroll
    for (int kb = 0; kb < 6; ++kb)
      acc = bwmma(ldfrag(arow, kb * 32, half), ldfrag(brow, kb * 32, half), acc);

    int   j   = n0 + lm;
    float bv  = qkvb[j];
    int   qi  = j / DIMC, rem = j % DIMC, hd = rem >> 5, d = rem & 31;
    unsigned short* ob = qkv_out + (((long)blk * 3 + qi) * HEADS + hd) * (NTOK * HD) + d;
    #pragma unroll
    for (int r = 0; r < 8; ++r) {
      int mrow = m0 + r + 8 * half;
      ob[(long)mrow * HD] = f2bf(acc[r] + bv);
    }
  }
}

// ---------------------------------------------------------------------------
// Kernel B: window attention, one block per (lonG, window, head).
// S = Q K^T * hd^-0.5 + earth-bias + shift-mask  (all recomputed in-kernel),
// softmax, O = P V. Dynamic LDS ~163 KB (CDNA5 allows 320 KB per workgroup).
// Q/K/V staged with async global->LDS b128 copies (ASYNCcnt).
// ---------------------------------------------------------------------------
__global__ __launch_bounds__(256) void k_attn(
    const unsigned short* __restrict__ qkv,
    const float* __restrict__ btab,
    unsigned short* __restrict__ att_out)
{
  extern __shared__ unsigned char smem[];
  unsigned short* Qs = (unsigned short*)smem;             // [144][32]
  unsigned short* Ks = Qs + NTOK * HD;                    // [144][32]
  unsigned short* Vt = Ks + NTOK * HD;                    // [32][160] transposed V
  float*          S  = (float*)(Vt + HD * 160);           // [144][160] fp32 scores
  unsigned short* P  = (unsigned short*)(S + NTOK * 160); // [144][160] bf16 probs
  unsigned short* Vraw = P;                               // [144][32], overlays P

  const int blk  = blockIdx.x;                            // 0..5759
  const int head = blk % HEADS;
  const int wwin = (blk / HEADS) % NWIN;
  const int lonG = blk / (HEADS * NWIN);
  const int wave = threadIdx.x >> 5, lane = threadIdx.x & 31;
  const int half = lane >> 4, lm = lane & 15;

  const long base = ((long)(lonG * NWIN + wwin) * 3 * HEADS + head) * (NTOK * HD);
  const unsigned short* Qg = qkv + base;
  const unsigned short* Kg = qkv + base + (long)HEADS * NTOK * HD;
  const unsigned short* Vg = qkv + base + (long)2 * HEADS * NTOK * HD;

  // ---- async stage Q, K, V(raw): 576 x b128 per matrix ----
  for (int i = threadIdx.x; i < NTOK * HD / 8; i += 256) {
    async_copy_b128((const u32x4*)Qg + i, (u32x4*)Qs + i);
    async_copy_b128((const u32x4*)Kg + i, (u32x4*)Ks + i);
    async_copy_b128((const u32x4*)Vg + i, (u32x4*)Vraw + i);
  }
  wait_async();
  __syncthreads();

  // ---- transpose V in LDS, zero K-pad to 160 ----
  for (int i = threadIdx.x; i < HD * 160; i += 256) {
    int d = i / 160, k = i % 160;
    Vt[d * 160 + k] = (k < NTOK) ? Vraw[k * HD + d] : (unsigned short)0;
  }
  __syncthreads();

  // ---- S tiles: 9x9, one WMMA each (K = HD = 32) ----
  const float scale = 0.17677669529663687f;               // 32^-0.5
  for (int tile = wave; tile < 81; tile += 8) {
    int mi = tile % 9, nj = tile / 9;
    int m0 = mi * 16, n0 = nj * 16;
    v8f acc = bwmma(ldfrag(Qs + (m0 + lm) * HD, 0, half),
                    ldfrag(Ks + (n0 + lm) * HD, 0, half), vzero8());
    int kcol = n0 + lm;
    int jz = kcol / 72, jh = (kcol / 12) % 6, jw = kcol % 12;
    int regK = region_id(wwin, lonG, kcol);
    #pragma unroll
    for (int r = 0; r < 8; ++r) {
      int qrow = m0 + r + 8 * half;
      int iz = qrow / 72, ih = (qrow / 12) % 6, iw = qrow % 12;
      int idx = (iz + 2 * jz) * 828 + (ih + 6 * jh) * 23 + (iw - jw + 11);
      float bias = btab[((long)idx * NWIN + wwin) * HEADS + head];
      float msk  = (region_id(wwin, lonG, qrow) == regK) ? 0.f : -100.f;
      S[qrow * 160 + kcol] = acc[r] * scale + bias + msk;
    }
  }
  __syncthreads();

  // ---- softmax: wave-per-row, shfl reductions; writes P (overwrites Vraw) ----
  for (int n = wave; n < NTOK; n += 8) {
    const float* srow = S + n * 160;
    float mx = -1e30f;
    for (int m = lane; m < NTOK; m += 32) mx = fmaxf(mx, srow[m]);
    #pragma unroll
    for (int m = 16; m; m >>= 1) mx = fmaxf(mx, __shfl_xor(mx, m, 32));
    float sum = 0.f;
    for (int m = lane; m < NTOK; m += 32) sum += __expf(srow[m] - mx);
    #pragma unroll
    for (int m = 16; m; m >>= 1) sum += __shfl_xor(sum, m, 32);
    float inv = 1.f / sum;
    unsigned short* prow = P + n * 160;
    for (int m = lane; m < 160; m += 32)
      prow[m] = (m < NTOK) ? f2bf(__expf(srow[m] - mx) * inv) : (unsigned short)0;
  }
  __syncthreads();

  // ---- O = P(144x160) * V(160x32): 9x2 tiles, K-loop of 5 ----
  for (int tile = wave; tile < 18; tile += 8) {
    int mi = tile % 9, nj = tile / 9;
    int m0 = mi * 16, n0 = nj * 16;
    v8f acc = vzero8();
    const unsigned short* arow = P  + (m0 + lm) * 160;
    const unsigned short* brow = Vt + (n0 + lm) * 160;    // B[k][n] = Vt[n][k]
    #pragma unroll
    for (int kb = 0; kb < 5; ++kb)
      acc = bwmma(ldfrag(arow, kb * 32, half), ldfrag(brow, kb * 32, half), acc);
    unsigned short* ob = att_out + ((long)(lonG * NWIN + wwin) * NTOK) * DIMC
                       + head * HD + n0 + lm;
    #pragma unroll
    for (int r = 0; r < 8; ++r) {
      int qrow = m0 + r + 8 * half;
      ob[(long)qrow * DIMC] = f2bf(acc[r]);
    }
  }
}

// ---------------------------------------------------------------------------
// Kernel C: proj GEMM + inverse window/roll scatter + residual -> x1 (fp32)
// ---------------------------------------------------------------------------
__global__ __launch_bounds__(256) void k_proj(
    const unsigned short* __restrict__ att,
    const unsigned short* __restrict__ wp_bf,
    const float* __restrict__ projb,
    const float* __restrict__ x,
    float* __restrict__ x1)
{
  __shared__ unsigned short Ah[NTOK * DIMC];
  const int blk  = blockIdx.x;
  const int lonG = blk / NWIN, w = blk % NWIN;
  const int wave = threadIdx.x >> 5, lane = threadIdx.x & 31;
  const int half = lane >> 4, lm = lane & 15;

  const unsigned short* ag = att + (long)blk * NTOK * DIMC;
  for (int i = threadIdx.x; i < NTOK * DIMC / 8; i += 256)
    async_copy_b128((const u32x4*)ag + i, (u32x4*)Ah + i);
  wait_async();
  __syncthreads();

  for (int tile = wave; tile < 9 * 12; tile += 8) {
    int mi = tile % 9, nj = tile / 9;
    int m0 = mi * 16, n0 = nj * 16;
    v8f acc = vzero8();
    const unsigned short* arow = Ah    + (m0 + lm) * DIMC;
    const unsigned short* brow = wp_bf + (n0 + lm) * DIMC;
    #pragma unroll
    for (int kb = 0; kb < 6; ++kb)
      acc = bwmma(ldfrag(arow, kb * 32, half), ldfrag(brow, kb * 32, half), acc);

    int   j  = n0 + lm;
    float pb = projb[j];
    #pragma unroll
    for (int r = 0; r < 8; ++r) {
      int n  = m0 + r + 8 * half;
      int wp = n / 72, wa = (n / 12) % 6, wo = n % 12;
      int plw  = (w >> 4) * 2 + wp;
      int latw = (w & 15) * 6 + wa;
      int lonw = lonG * 12 + wo;
      int plp  = (plw + 1) & 7;
      int latp = latw + 3; if (latp >= 96)  latp -= 96;
      int lon  = lonw + 6; if (lon  >= 180) lon  -= 180;
      int lat  = latp - 2;
      if (lat >= 0 && lat < 91) {
        long t = ((long)plp * 91 + lat) * 180 + lon;
        x1[t * DIMC + j] = x[t * DIMC + j] + acc[r] + pb;
      }
    }
  }
}

// ---------------------------------------------------------------------------
// Kernel D: LN2 + FC1 (192->768) + exact GELU -> mid (bf16)
// ---------------------------------------------------------------------------
__global__ __launch_bounds__(256) void k_mlp1(
    const float* __restrict__ x1,
    const float* __restrict__ n2w, const float* __restrict__ n2b,
    const unsigned short* __restrict__ w1, const float* __restrict__ b1,
    unsigned short* __restrict__ mid)
{
  __shared__ unsigned short Ah[NTOK * DIMC];
  const long t0   = (long)blockIdx.x * NTOK;
  const int  wave = threadIdx.x >> 5, lane = threadIdx.x & 31;
  const int  half = lane >> 4, lm = lane & 15;

  for (int n = wave; n < NTOK; n += 8) {
    const float* xr = x1 + (t0 + n) * DIMC;
    float vals[6], s = 0.f, s2 = 0.f;
    #pragma unroll
    for (int i = 0; i < 6; ++i) { float v = xr[lane + 32 * i]; vals[i] = v; s += v; s2 += v * v; }
    #pragma unroll
    for (int m = 16; m; m >>= 1) { s += __shfl_xor(s, m, 32); s2 += __shfl_xor(s2, m, 32); }
    float mean = s * (1.f / DIMC);
    float inv  = rsqrtf(s2 * (1.f / DIMC) - mean * mean + 1e-5f);
    #pragma unroll
    for (int i = 0; i < 6; ++i) {
      int kk = lane + 32 * i;
      Ah[n * DIMC + kk] = f2bf((vals[i] - mean) * inv * n2w[kk] + n2b[kk]);
    }
  }
  __syncthreads();

  for (int tile = wave; tile < 9 * 48; tile += 8) {
    int mi = tile % 9, nj = tile / 9;
    int m0 = mi * 16, n0 = nj * 16;
    v8f acc = vzero8();
    const unsigned short* arow = Ah + (m0 + lm) * DIMC;
    const unsigned short* brow = w1 + (n0 + lm) * DIMC;
    #pragma unroll
    for (int kb = 0; kb < 6; ++kb)
      acc = bwmma(ldfrag(arow, kb * 32, half), ldfrag(brow, kb * 32, half), acc);

    int   j  = n0 + lm;
    float bb = b1[j];
    #pragma unroll
    for (int r = 0; r < 8; ++r) {
      float v = acc[r] + bb;
      float g = 0.5f * v * (1.f + erff(v * 0.70710678118654752f));
      mid[(t0 + m0 + r + 8 * half) * HIDDEN + j] = f2bf(g);
    }
  }
}

// ---------------------------------------------------------------------------
// Kernel E: FC2 (768->192) + residual -> out (fp32)
// ---------------------------------------------------------------------------
__global__ __launch_bounds__(256) void k_mlp2(
    const unsigned short* __restrict__ mid,
    const unsigned short* __restrict__ w2, const float* __restrict__ b2,
    const float* __restrict__ x1,
    float* __restrict__ out)
{
  const long t0   = (long)blockIdx.x * NTOK;
  const int  wave = threadIdx.x >> 5, lane = threadIdx.x & 31;
  const int  half = lane >> 4, lm = lane & 15;

  for (int tile = wave; tile < 9 * 12; tile += 8) {
    int mi = tile % 9, nj = tile / 9;
    int m0 = mi * 16, n0 = nj * 16;
    v8f acc = vzero8();
    const unsigned short* arow = mid + (t0 + m0 + lm) * HIDDEN;   // L2-resident bf16
    const unsigned short* brow = w2  + (n0 + lm) * HIDDEN;
    #pragma unroll
    for (int kb = 0; kb < 24; ++kb)
      acc = bwmma(ldfrag(arow, kb * 32, half), ldfrag(brow, kb * 32, half), acc);

    int   j  = n0 + lm;
    float bb = b2[j];
    #pragma unroll
    for (int r = 0; r < 8; ++r) {
      long t = t0 + m0 + r + 8 * half;
      out[t * DIMC + j] = x1[t * DIMC + j] + acc[r] + bb;
    }
  }
}

// ---------------------------------------------------------------------------
extern "C" void kernel_launch(void* const* d_in, const int* in_sizes, int n_in,
                              void* d_out, int out_size, void* d_ws, size_t ws_size,
                              hipStream_t stream)
{
  (void)in_sizes; (void)n_in; (void)out_size; (void)ws_size;
  const float* x    = (const float*)d_in[0];
  const float* n1w  = (const float*)d_in[1];
  const float* n1b  = (const float*)d_in[2];
  const float* qkvw = (const float*)d_in[3];
  const float* qkvb = (const float*)d_in[4];
  const float* btab = (const float*)d_in[5];
  const float* pw   = (const float*)d_in[6];
  const float* pb   = (const float*)d_in[7];
  const float* n2w  = (const float*)d_in[8];
  const float* n2b  = (const float*)d_in[9];
  const float* w1   = (const float*)d_in[10];
  const float* b1   = (const float*)d_in[11];
  const float* w2   = (const float*)d_in[12];
  const float* b2   = (const float*)d_in[13];
  float* out = (float*)d_out;
  char*  ws  = (char*)d_ws;

  size_t off = 0;
  auto take = [&](size_t b) { size_t r = off; off += (b + 255) & ~(size_t)255; return r; };
  unsigned short* wqb = (unsigned short*)(ws + take((size_t)3 * DIMC * DIMC * 2)); // 576x192
  unsigned short* wpb = (unsigned short*)(ws + take((size_t)DIMC * DIMC * 2));     // 192x192
  unsigned short* w1b = (unsigned short*)(ws + take((size_t)HIDDEN * DIMC * 2));   // 768x192
  unsigned short* w2b = (unsigned short*)(ws + take((size_t)DIMC * HIDDEN * 2));   // 192x768
  float*          x1  = (float*)(ws + take((size_t)LTOK * DIMC * 4));

  // big region: qkv (159 MB) + att (53 MB) live during kernels A-C;
  // mid (201 MB) aliases the same region afterwards.
  size_t qkv_bytes = ((size_t)NBLK * 3 * HEADS * NTOK * HD * 2 + 255) & ~(size_t)255;
  size_t big       = off;
  unsigned short* qkvbuf = (unsigned short*)(ws + big);
  unsigned short* attbuf = (unsigned short*)(ws + big + qkv_bytes);
  unsigned short* midbuf = (unsigned short*)(ws + big);   // alias: qkv dead by kernel D

  k_cvt<<<(3 * DIMC * DIMC + 255) / 256, 256, 0, stream>>>(qkvw, wqb, 3 * DIMC * DIMC);
  k_cvt<<<(DIMC * DIMC + 255) / 256,     256, 0, stream>>>(pw,   wpb, DIMC * DIMC);
  k_cvt<<<(HIDDEN * DIMC + 255) / 256,   256, 0, stream>>>(w1,   w1b, HIDDEN * DIMC);
  k_cvt<<<(DIMC * HIDDEN + 255) / 256,   256, 0, stream>>>(w2,   w2b, DIMC * HIDDEN);

  k_ln1_qkv<<<NBLK, 256, 0, stream>>>(x, n1w, n1b, wqb, qkvb, qkvbuf);

  const size_t attn_lds = (size_t)NTOK * HD * 2 * 2   // Qs + Ks
                        + (size_t)HD * 160 * 2        // Vt
                        + (size_t)NTOK * 160 * 4      // S (fp32)
                        + (size_t)NTOK * 160 * 2;     // P (bf16, overlays Vraw)
  k_attn<<<NBLK * HEADS, 256, attn_lds, stream>>>(qkvbuf, btab, attbuf);

  k_proj<<<NBLK, 256, 0, stream>>>(attbuf, wpb, pb, x, x1);
  k_mlp1<<<TBLK, 256, 0, stream>>>(x1, n2w, n2b, w1b, b1, midbuf);
  k_mlp2<<<TBLK, 256, 0, stream>>>(midbuf, w2b, b2, x1, out);
}
